// RNN_39419209843019
// MI455X (gfx1250) — matrix operand order repaired and checked by reading, not compiled, not run on hardware
//
#include <hip/hip_runtime.h>

// ---------------------------------------------------------------------------
// Types for CDNA5 WMMA (wave32): V_WMMA_F32_16X16X32_BF16
// ---------------------------------------------------------------------------
typedef __bf16 v16bf __attribute__((ext_vector_type(16)));
typedef float  v8f   __attribute__((ext_vector_type(8)));
typedef unsigned short v16us __attribute__((ext_vector_type(16)));

union V16 {
    v16us u;
    v16bf b;
};

static __device__ __forceinline__ unsigned short f32_to_bf16_hw(float f) {
    return __builtin_bit_cast(unsigned short, (__bf16)f);   // v_cvt_*bf16_f32
}

// Cheap tanh: tanh(x) = (1 - e^{-2x}) / (1 + e^{-2x}), e^{-2x} via v_exp_f32.
static __device__ __forceinline__ float fast_tanh(float x) {
    float xc = fminf(9.0f, fmaxf(-9.0f, x));
    float q  = exp2f(xc * -2.8853900817779268f);   // exp(-2x)
    return (1.0f - q) / (1.0f + q);
}

// Problem constants
#define RNN_B    32
#define RNN_T    2048
#define RNN_D    512      // D_IN == D_OUT == 512

// ---------------------------------------------------------------------------
// Kernel 1: pack W and W_rec (f32 [512,512], k-major) into the WMMA
// B-operand lane layout, bf16.
// Tile (kc, nt): 32 K x 16 N.  Lane L: n = nt*16 + (L&15),
//   lanes 0-15 hold K = kc*32 + (j&7) + (j>=8 ? 16 : 0) (j=0..15), lanes 16-31 hold that K + 8.
// Packed element index: ((kc*32+nt)*32 + lane)*16 + j
// ---------------------------------------------------------------------------
__global__ __launch_bounds__(256) void pack_weights(
    const float* __restrict__ W,
    const float* __restrict__ Wrec,
    unsigned short* __restrict__ Wpack,
    unsigned short* __restrict__ Wrecpack)
{
    const int PER = RNN_D * RNN_D;                 // 262144 elements / matrix
    int tid = blockIdx.x * blockDim.x + threadIdx.x;
    int sel = tid >= PER;
    int e   = sel ? tid - PER : tid;

    int j    = e & 15;
    int lane = (e >> 4) & 31;
    int tile = e >> 9;            // = kc*32 + nt
    int kc   = tile >> 5;
    int nt   = tile & 31;

    int n = nt * 16 + (lane & 15);
    int k = kc * 32 + ((lane & 16) ? 8 : 0) + (j & 7) + ((j & 8) ? 16 : 0);

    const float* src = sel ? Wrec : W;
    unsigned short v = f32_to_bf16_hw(src[k * RNN_D + n]);
    (sel ? Wrecpack : Wpack)[e] = v;
}

// ---------------------------------------------------------------------------
// Kernel 2: xw[t][b][:] = x[b][t][:] @ W + bias   (M = T*B rows, r = t*32 + b)
// Each wave computes a 16x64 strip (4 N-tiles): one A fragment feeds 4 WMMAs.
// A fragment (16x32 bf16), converted in-register with HW bf16 casts:
//   lane<16:  e[0..7]=K0..7,  e[8..15]=K16..23   (M = lane&15)
//   lane>=16: e[0..7]=K8..15, e[8..15]=K24..31
// ---------------------------------------------------------------------------
__global__ __launch_bounds__(256) void input_gemm(
    const float* __restrict__ x,                // [32][2048][512]
    const unsigned short* __restrict__ Wpack,   // packed bf16 B tiles
    const float* __restrict__ bias,             // [512]
    float* __restrict__ xw)                     // [2048][32][512]
{
    const int lane = threadIdx.x & 31;
    const int wave = threadIdx.x >> 5;
    const int job  = blockIdx.x * 8 + wave;     // 0 .. 32767
    const int mt = job >> 3;                    // 0 .. 4095  (16-row tile)
    const int ns = job & 7;                     // 0 .. 7     (64-col strip)
    const int m0 = mt << 4;
    const int n0 = ns << 6;

    // This lane's A row: r = t*32 + b  ->  b = r&31, t = r>>5
    const int r  = m0 + (lane & 15);
    const int bb = r & 31;
    const int tt = r >> 5;
    const float* xrow = x + ((size_t)bb * RNN_T + tt) * RNN_D;
    const int kb = (lane & 16) ? 8 : 0;

    v8f c0 = {}, c1 = {}, c2 = {}, c3 = {};
    #pragma unroll 2
    for (int kc = 0; kc < 16; ++kc) {
        const float* xk = xrow + kc * 32;
        float4 f0 = *(const float4*)(xk + kb);
        float4 f1 = *(const float4*)(xk + kb + 4);
        float4 f2 = *(const float4*)(xk + 16 + kb);
        float4 f3 = *(const float4*)(xk + 16 + kb + 4);

        v16bf a;
        a[0]  = (__bf16)f0.x;  a[1]  = (__bf16)f0.y;
        a[2]  = (__bf16)f0.z;  a[3]  = (__bf16)f0.w;
        a[4]  = (__bf16)f1.x;  a[5]  = (__bf16)f1.y;
        a[6]  = (__bf16)f1.z;  a[7]  = (__bf16)f1.w;
        a[8]  = (__bf16)f2.x;  a[9]  = (__bf16)f2.y;
        a[10] = (__bf16)f2.z;  a[11] = (__bf16)f2.w;
        a[12] = (__bf16)f3.x;  a[13] = (__bf16)f3.y;
        a[14] = (__bf16)f3.z;  a[15] = (__bf16)f3.w;

        const unsigned short* wrow =
            Wpack + ((((size_t)kc * 32 + ns * 4) * 32 + lane) << 4);
        V16 b0, b1, b2, b3;
        b0.u = *(const v16us*)(wrow);            // nt = ns*4 + 0
        b1.u = *(const v16us*)(wrow + 512);      // nt = ns*4 + 1  (tile = 512 us)
        b2.u = *(const v16us*)(wrow + 1024);     // nt = ns*4 + 2
        b3.u = *(const v16us*)(wrow + 1536);     // nt = ns*4 + 3

        c0 = __builtin_amdgcn_wmma_f32_16x16x32_bf16(false, a, false, b0.b, (short)0, c0, false, false);
        c1 = __builtin_amdgcn_wmma_f32_16x16x32_bf16(false, a, false, b1.b, (short)0, c1, false, false);
        c2 = __builtin_amdgcn_wmma_f32_16x16x32_bf16(false, a, false, b2.b, (short)0, c2, false, false);
        c3 = __builtin_amdgcn_wmma_f32_16x16x32_bf16(false, a, false, b3.b, (short)0, c3, false, false);
    }

    // C/D layout: VGPR j -> row j (lanes 0-15) / row 8+j (lanes 16-31), col = lane&15
    const int rowofs = (lane & 16) ? 8 : 0;
    v8f cc[4] = {c0, c1, c2, c3};
    #pragma unroll
    for (int u = 0; u < 4; ++u) {
        const int n = n0 + u * 16 + (lane & 15);
        const float bv = bias[n];
        #pragma unroll
        for (int j = 0; j < 8; ++j) {
            int rr = m0 + rowofs + j;
            xw[(size_t)rr * RNN_D + n] = cc[u][j] + bv;
        }
    }
}

// ---------------------------------------------------------------------------
// Kernel 3: persistent single-WGP recurrent scan.
// 512 threads = 16 waves (4/SIMD); wave w owns columns [w*32, w*32+32)
// (N-tiles 2w, 2w+1).  Per step each wave reads full h from LDS once
// (LDS h-traffic = 16 waves * 32KB = 512KB/step, balanced vs ~1024 WMMA).
// h kept in LDS pre-packed in the WMMA A lane layout:
//   hA[mt][kc][lane][j]  (mt = row/16, kc = K/32), 32 KB total.
// W_rec B-fragments for kc 0..7 are held in registers across all timesteps
// (loop-invariant, 128 VGPRs/lane); kc 8..15 streamed from L2-resident pack.
// ---------------------------------------------------------------------------
__global__ __launch_bounds__(512) void rnn_scan(
    const float* __restrict__ xw,                 // [2048][32][512]
    const unsigned short* __restrict__ Wrecpack,  // packed bf16 B tiles
    float* __restrict__ out)                      // [32][2048][512]
{
    __shared__ __align__(32) unsigned short hA[2][16][32][16];  // 32 KB

    const int lane = threadIdx.x & 31;
    const int w    = threadIdx.x >> 5;            // 0..15
    const int rowofs = (lane & 16) ? 8 : 0;

    // h0 = 0
    {
        unsigned short* p = &hA[0][0][0][0];
        for (int i = threadIdx.x; i < 2 * 16 * 32 * 16; i += 512) p[i] = 0;
    }

    // Hoist W_rec B fragments for kc = 0..7, both owned N-tiles (nt = 2w, 2w+1)
    V16 Breg[2][8];
    #pragma unroll
    for (int u = 0; u < 2; ++u)
        #pragma unroll
        for (int kc = 0; kc < 8; ++kc)
            Breg[u][kc].u = *(const v16us*)(Wrecpack +
                ((((size_t)kc * 32 + 2 * w + u) * 32 + lane) << 4));

    // Per-column constants and inverse A-layout scatter map (K = n next step)
    int ncol[2], kcN[2], laneA[2], jA[2];
    #pragma unroll
    for (int u = 0; u < 2; ++u) {
        int n = w * 32 + u * 16 + (lane & 15);
        ncol[u]  = n;
        kcN[u]   = n >> 5;
        int kl   = n & 31;
        laneA[u] = (kl & 8) ? 16 : 0;                // + (row & 15)
        jA[u]    = (kl & 7) | ((kl & 16) >> 1);      // +8 if K in upper half
    }

    __syncthreads();

    for (int t = 0; t < RNN_T; ++t) {
        v8f c00 = {}, c01 = {}, c10 = {}, c11 = {};  // c[mt][u]

        #pragma unroll
        for (int kc = 0; kc < 8; ++kc) {             // B from registers
            V16 a0, a1;
            a0.u = *(const v16us*)&hA[0][kc][lane][0];
            a1.u = *(const v16us*)&hA[1][kc][lane][0];
            c00 = __builtin_amdgcn_wmma_f32_16x16x32_bf16(false, a0.b, false, Breg[0][kc].b, (short)0, c00, false, false);
            c10 = __builtin_amdgcn_wmma_f32_16x16x32_bf16(false, a1.b, false, Breg[0][kc].b, (short)0, c10, false, false);
            c01 = __builtin_amdgcn_wmma_f32_16x16x32_bf16(false, a0.b, false, Breg[1][kc].b, (short)0, c01, false, false);
            c11 = __builtin_amdgcn_wmma_f32_16x16x32_bf16(false, a1.b, false, Breg[1][kc].b, (short)0, c11, false, false);
        }
        #pragma unroll 2
        for (int kc = 8; kc < 16; ++kc) {            // B streamed (L2-hot)
            const unsigned short* wrow = Wrecpack +
                ((((size_t)kc * 32 + 2 * w) * 32 + lane) << 4);
            V16 b0, b1;
            b0.u = *(const v16us*)(wrow);
            b1.u = *(const v16us*)(wrow + 512);
            V16 a0, a1;
            a0.u = *(const v16us*)&hA[0][kc][lane][0];
            a1.u = *(const v16us*)&hA[1][kc][lane][0];
            c00 = __builtin_amdgcn_wmma_f32_16x16x32_bf16(false, a0.b, false, b0.b, (short)0, c00, false, false);
            c10 = __builtin_amdgcn_wmma_f32_16x16x32_bf16(false, a1.b, false, b0.b, (short)0, c10, false, false);
            c01 = __builtin_amdgcn_wmma_f32_16x16x32_bf16(false, a0.b, false, b1.b, (short)0, c01, false, false);
            c11 = __builtin_amdgcn_wmma_f32_16x16x32_bf16(false, a1.b, false, b1.b, (short)0, c11, false, false);
        }
        __syncthreads();   // everyone done reading h_{t-1} from LDS

        const float* xwt = xw + (size_t)t * (RNN_B * RNN_D);
        if (t + 1 < RNN_T) {
            // global_prefetch_b8: pull next step's xw tile toward the caches
            __builtin_prefetch(xw + (size_t)(t + 1) * (RNN_B * RNN_D)
                                  + (size_t)threadIdx.x * 32, 0, 0);
        }

        v8f cm0[2] = {c00, c01};   // mt 0, u = 0,1
        v8f cm1[2] = {c10, c11};   // mt 1, u = 0,1
        #pragma unroll
        for (int u = 0; u < 2; ++u) {
            const int n = ncol[u];
            #pragma unroll
            for (int j = 0; j < 8; ++j) {
                {   // M-tile 0: rows 0..15
                    int row = rowofs + j;
                    float v = fast_tanh(cm0[u][j] + xwt[row * RNN_D + n]);
                    out[(size_t)row * (RNN_T * RNN_D) + (size_t)t * RNN_D + n] = v;
                    hA[0][kcN[u]][(row & 15) + laneA[u]][jA[u]] = f32_to_bf16_hw(v);
                }
                {   // M-tile 1: rows 16..31
                    int row = 16 + rowofs + j;
                    float v = fast_tanh(cm1[u][j] + xwt[row * RNN_D + n]);
                    out[(size_t)row * (RNN_T * RNN_D) + (size_t)t * RNN_D + n] = v;
                    hA[1][kcN[u]][(row & 15) + laneA[u]][jA[u]] = f32_to_bf16_hw(v);
                }
            }
        }
        __syncthreads();   // new h_t visible to all waves
    }
}

// ---------------------------------------------------------------------------
// Host launcher
// ---------------------------------------------------------------------------
extern "C" void kernel_launch(void* const* d_in, const int* in_sizes, int n_in,
                              void* d_out, int out_size, void* d_ws, size_t ws_size,
                              hipStream_t stream) {
    (void)in_sizes; (void)n_in; (void)out_size; (void)ws_size;

    const float* x    = (const float*)d_in[0];   // [32][2048][512]
    const float* W    = (const float*)d_in[1];   // [512][512]
    const float* Wrec = (const float*)d_in[2];   // [512][512]
    const float* bias = (const float*)d_in[3];   // [512]
    float* out = (float*)d_out;                  // [32][2048][512]

    // Workspace layout:
    //   [0, 134217728)               : xw, f32 [2048][32][512]
    //   [134217728, +524288)         : Wpack    (bf16, B-operand layout)
    //   [134742016, +524288)         : Wrecpack (bf16, B-operand layout)
    char* ws = (char*)d_ws;
    float* xw = (float*)ws;
    unsigned short* Wpack    = (unsigned short*)(ws + 134217728u);
    unsigned short* Wrecpack = (unsigned short*)(ws + 134217728u + 524288u);

    // 2 * 262144 elements / 256 threads
    pack_weights<<<2048, 256, 0, stream>>>(W, Wrec, Wpack, Wrecpack);

    // 4096 M-tiles * 8 N-strips = 32768 wave jobs / 8 waves per block
    input_gemm<<<4096, 256, 0, stream>>>(x, Wpack, bias, xw);

    // Persistent single-workgroup scan (16 waves, 4 per SIMD, on one WGP)
    rnn_scan<<<1, 512, 0, stream>>>(xw, Wrecpack, out);
}
